// perceiver_23390391894848
// MI455X (gfx1250) — matrix operand order repaired and checked by reading, not compiled, not run on hardware
//
#include <hip/hip_runtime.h>
#include <hip/hip_bf16.h>
#include <math.h>

// ---------------------------------------------------------------------------
// Perceiver forward for MI455X (gfx1250).
// All matmuls: bf16 WMMA (v_wmma_f32_16x16x32_bf16), f32 accumulate.
// Every GEMM is "NT" (B stored [N][K] row-major): A and B fragments are each
// two ds_load_b128. LDS is double-buffered; next tile's global loads are
// issued before the WMMAs and the dependent LDS stores after them, so global
// latency is hidden behind the matrix pipe.
// ---------------------------------------------------------------------------

typedef __bf16 bf16;
typedef __attribute__((ext_vector_type(16))) __bf16 bf16x16;
typedef __attribute__((ext_vector_type(8)))  float  f32x8;

#define D_MODEL 1024
#define NBATCH  16
#define SEQ     2048
#define LLAT    256
#define NOUT    5

__device__ __forceinline__ float mish_f(float x) {
  float sp = (x > 20.0f) ? x : log1pf(__expf(x));   // softplus
  return x * tanhf(sp);
}

// ---------------- elementwise kernels ----------------

__global__ void pe_add_cvt(const float* __restrict__ x, bf16* __restrict__ xp, int total) {
  int idx = blockIdx.x * blockDim.x + threadIdx.x;
  if (idx >= total) return;
  int d = idx & (D_MODEL - 1);
  int l = (idx >> 10) & (SEQ - 1);
  int i2 = d & ~1;
  float freq = __expf(-9.210340371976184f * (float)i2 / (float)D_MODEL); // exp(-ln(1e4)*i2/D)
  float ang  = (float)l * freq;
  float pe   = (d & 1) ? __cosf(ang) : __sinf(ang);
  xp[idx] = (bf16)(x[idx] + pe);
}

// Wt[n][k] = bf16(W[k][n])   (DxD square)
__global__ void cvt_tr_f32_bf16(const float* __restrict__ in, bf16* __restrict__ out) {
  int idx = blockIdx.x * blockDim.x + threadIdx.x;
  if (idx >= D_MODEL * D_MODEL) return;
  int n = idx >> 10, k = idx & (D_MODEL - 1);
  out[idx] = (bf16)in[k * D_MODEL + n];
}

__global__ void fill_bf16_zero(bf16* __restrict__ p, int n) {
  int i = blockIdx.x * blockDim.x + threadIdx.x;
  if (i < n) p[i] = (bf16)0.0f;
}

// ---------------- bf16 WMMA GEMM (NT form, LDS double-buffered) ----------------
// C[M,N] = act( scale * (A[M,K] @ B^T) + bias ),  B stored [N,K] row-major.
// Batched via blockIdx.z (element strides sA/sB/sC; 0 => shared operand).
// Output: f32 if Cf != null else bf16; trans_store => write C^T (ldc = rowstride of C^T).
// Block: 128 threads = 4 waves; block tile 128x64; wave tile 64x32 (4x2 WMMA).
#define LDSA 40                       // row stride (elems): 16B aligned, bank-spread
#define ABUF (128 * LDSA)             // A region size per buffer
#define BUFE ((128 + 64) * LDSA)      // one buffer: A tile + B tile

__global__ __launch_bounds__(128, 1)
void gemm_bf16_nt(const bf16* __restrict__ A, const bf16* __restrict__ B,
                  const float* __restrict__ bias,
                  bf16* __restrict__ Cb, float* __restrict__ Cf,
                  int M, int N, int K, int lda, int ldb, int ldc,
                  long long sA, long long sB, long long sC,
                  float scale, int act, int trans_store)
{
  __shared__ __align__(16) bf16 smem[2 * BUFE];

  const int tid  = threadIdx.x;
  const int bz   = blockIdx.z;
  const int m0   = blockIdx.y * 128;
  const int n0   = blockIdx.x * 64;
  const bf16* __restrict__ Ab = A + (size_t)bz * (size_t)sA;
  const bf16* __restrict__ Bb = B + (size_t)bz * (size_t)sB;

  const int wave = tid >> 5;
  const int lane = tid & 31;
  const int half = lane >> 4;          // 0: lanes 0-15, 1: lanes 16-31
  const int mr   = lane & 15;
  const int wm   = (wave >> 1) * 64;   // wave tile origin in block tile
  const int wn   = (wave & 1) * 32;

  // staging geometry: row = (tid>>2) + 32j, col = (tid&3)*8 (j-invariant)
  const int r0 = tid >> 2;             // 0..31
  const int c0 = (tid & 3) << 3;       // 0,8,16,24
  const bf16* __restrict__ ag = Ab + (size_t)(m0 + r0) * lda + c0;
  const bf16* __restrict__ bg = Bb + (size_t)(n0 + r0) * ldb + c0;
  const size_t a32 = (size_t)32 * lda;
  const size_t b32 = (size_t)32 * ldb;
  const int aw = r0 * LDSA + c0;            // LDS write offsets (elements)
  const int bw = ABUF + r0 * LDSA + c0;

  // ---- stage first K-slice into buffer 0 ----
  {
    uint4 t0 = *(const uint4*)(ag);
    uint4 t1 = *(const uint4*)(ag + a32);
    uint4 t2 = *(const uint4*)(ag + 2 * a32);
    uint4 t3 = *(const uint4*)(ag + 3 * a32);
    uint4 t4 = *(const uint4*)(bg);
    uint4 t5 = *(const uint4*)(bg + b32);
    *(uint4*)(smem + aw)             = t0;
    *(uint4*)(smem + aw + 32 * LDSA) = t1;
    *(uint4*)(smem + aw + 64 * LDSA) = t2;
    *(uint4*)(smem + aw + 96 * LDSA) = t3;
    *(uint4*)(smem + bw)             = t4;
    *(uint4*)(smem + bw + 32 * LDSA) = t5;
  }
  __syncthreads();

  f32x8 acc[4][2];
#pragma unroll
  for (int i = 0; i < 4; ++i)
#pragma unroll
    for (int j = 0; j < 2; ++j)
#pragma unroll
      for (int e = 0; e < 8; ++e) acc[i][j][e] = 0.0f;

  union Frag { bf16x16 v; uint4 q[2]; };

  int ping = 0;
  for (int k0 = 0; k0 < K; k0 += 32, ping ^= 1) {
    const bool pf = (k0 + 32) < K;

    // issue next-slice global loads early (wait happens at the stores below,
    // hidden behind the WMMA block)
    uint4 t0, t1, t2, t3, t4, t5;
    if (pf) {
      const bf16* agn = ag + (k0 + 32);
      const bf16* bgn = bg + (k0 + 32);
      t0 = *(const uint4*)(agn);
      t1 = *(const uint4*)(agn + a32);
      t2 = *(const uint4*)(agn + 2 * a32);
      t3 = *(const uint4*)(agn + 3 * a32);
      t4 = *(const uint4*)(bgn);
      t5 = *(const uint4*)(bgn + b32);
    }

    // ---- compute from current buffer ----
    const bf16* s = smem + ping * BUFE;
    // B fragments: elements [16h .. 16h+15] of rows (wn+nt*16+mr)
    Frag bf0, bf1;
    {
      const bf16* bp = s + ABUF + (wn + mr) * LDSA + 16 * half;
      bf0.q[0] = *(const uint4*)(bp);
      bf0.q[1] = *(const uint4*)(bp + 8);
      bf1.q[0] = *(const uint4*)(bp + 16 * LDSA);
      bf1.q[1] = *(const uint4*)(bp + 16 * LDSA + 8);
    }
#pragma unroll
    for (int mt = 0; mt < 4; ++mt) {
      // A fragment: elements [16h .. 16h+15] of row (wm+mt*16+mr)
      Frag af;
      const bf16* ap = s + (wm + mt * 16 + mr) * LDSA + 16 * half;
      af.q[0] = *(const uint4*)(ap);
      af.q[1] = *(const uint4*)(ap + 8);
      acc[mt][0] = __builtin_amdgcn_wmma_f32_16x16x32_bf16(
          false, af.v, false, bf0.v, (short)0, acc[mt][0], false, false);
      acc[mt][1] = __builtin_amdgcn_wmma_f32_16x16x32_bf16(
          false, af.v, false, bf1.v, (short)0, acc[mt][1], false, false);
    }

    // ---- store next slice into alternate buffer ----
    if (pf) {
      bf16* sw = smem + (ping ^ 1) * BUFE;
      *(uint4*)(sw + aw)             = t0;
      *(uint4*)(sw + aw + 32 * LDSA) = t1;
      *(uint4*)(sw + aw + 64 * LDSA) = t2;
      *(uint4*)(sw + aw + 96 * LDSA) = t3;
      *(uint4*)(sw + bw)             = t4;
      *(uint4*)(sw + bw + 32 * LDSA) = t5;
    }
    __syncthreads();
  }

  // ---- epilogue: scale, bias, activation, (optionally transposed) store ----
  bf16*  Cbb = Cb ? (Cb + (size_t)bz * (size_t)sC) : (bf16*)nullptr;
  float* Cfb = Cf ? (Cf + (size_t)bz * (size_t)sC) : (float*)nullptr;
#pragma unroll
  for (int mt = 0; mt < 4; ++mt)
#pragma unroll
    for (int nt = 0; nt < 2; ++nt)
#pragma unroll
      for (int e = 0; e < 8; ++e) {
        int row = m0 + wm + mt * 16 + e + (half ? 8 : 0);  // C layout: VGPR e -> M=e / e+8
        int col = n0 + wn + nt * 16 + mr;                  // N = lane&15
        float vv = acc[mt][nt][e] * scale;
        if (bias) vv += bias[col];
        if (act)  vv  = mish_f(vv);
        size_t off = trans_store ? ((size_t)col * ldc + row)
                                 : ((size_t)row * ldc + col);
        if (Cfb) Cfb[off] = vv;
        else     Cbb[off] = (bf16)vv;
      }
}

// ---------------- row softmax: f32 scores -> bf16 probs ----------------
__global__ __launch_bounds__(256)
void softmax_rows(const float* __restrict__ S, bf16* __restrict__ P, int cols) {
  __shared__ float red[256];
  int row = blockIdx.x;
  const float* s = S + (size_t)row * cols;
  bf16*        p = P + (size_t)row * cols;
  int t = threadIdx.x;

  float mx = -3.4e38f;
  for (int c = t; c < cols; c += 256) mx = fmaxf(mx, s[c]);
  red[t] = mx; __syncthreads();
  for (int w = 128; w > 0; w >>= 1) { if (t < w) red[t] = fmaxf(red[t], red[t + w]); __syncthreads(); }
  mx = red[0]; __syncthreads();

  float sum = 0.0f;
  for (int c = t; c < cols; c += 256) sum += __expf(s[c] - mx);
  red[t] = sum; __syncthreads();
  for (int w = 128; w > 0; w >>= 1) { if (t < w) red[t] += red[t + w]; __syncthreads(); }
  float inv = 1.0f / red[0];

  for (int c = t; c < cols; c += 256) p[c] = (bf16)(__expf(s[c] - mx) * inv);
}

// ---------------- head: mean over latent -> 1024x5 -> softmax ----------------
__global__ __launch_bounds__(256)
void head_kernel(const bf16* __restrict__ lat, const float* __restrict__ We,
                 const float* __restrict__ be, float* __restrict__ out) {
  __shared__ float avg[D_MODEL];
  __shared__ float logit[8];
  int b = blockIdx.x, t = threadIdx.x;
  const bf16* lb = lat + (size_t)b * LLAT * D_MODEL;
  for (int d = t; d < D_MODEL; d += 256) {
    float s = 0.0f;
    for (int l = 0; l < LLAT; ++l) s += (float)lb[l * D_MODEL + d];
    avg[d] = s * (1.0f / LLAT);
  }
  __syncthreads();
  if (t < NOUT) {
    float s = be[t];
    for (int d = 0; d < D_MODEL; ++d) s += avg[d] * We[d * NOUT + t];
    logit[t] = s;
  }
  __syncthreads();
  if (t == 0) {
    float mx = logit[0];
    for (int j = 1; j < NOUT; ++j) mx = fmaxf(mx, logit[j]);
    float e[NOUT], sum = 0.0f;
    for (int j = 0; j < NOUT; ++j) { e[j] = __expf(logit[j] - mx); sum += e[j]; }
    for (int j = 0; j < NOUT; ++j) out[b * NOUT + j] = e[j] / sum;
  }
}

// ---------------- host-side launch helper ----------------
static inline void gemm(hipStream_t st, const bf16* A, const bf16* B,
                        const float* bias, bf16* Cb, float* Cf,
                        int M, int N, int K, int lda, int ldb, int ldc,
                        long long sA, long long sB, long long sC,
                        int batch, float scale, int act, int trans_store) {
  dim3 g(N / 64, M / 128, batch), blk(128);
  gemm_bf16_nt<<<g, blk, 0, st>>>(A, B, bias, Cb, Cf, M, N, K, lda, ldb, ldc,
                                  sA, sB, sC, scale, act, trans_store);
}

extern "C" void kernel_launch(void* const* d_in, const int* in_sizes, int n_in,
                              void* d_out, int out_size, void* d_ws, size_t ws_size,
                              hipStream_t stream) {
  (void)in_sizes; (void)n_in; (void)out_size; (void)ws_size;

  const float* x   = (const float*)d_in[0];
  const float* cbq = (const float*)d_in[2];
  const float* cbk = (const float*)d_in[4];
  const float* cbv = (const float*)d_in[6];
  const float* cbo = (const float*)d_in[8];
  const float* sbq = (const float*)d_in[10];
  const float* sbk = (const float*)d_in[12];
  const float* sbv = (const float*)d_in[14];
  const float* sbo = (const float*)d_in[16];
  const float* d1b1 = (const float*)d_in[18];
  const float* d1b2 = (const float*)d_in[20];
  const float* d2b1 = (const float*)d_in[22];
  const float* d2b2 = (const float*)d_in[24];
  const float* We  = (const float*)d_in[25];
  const float* be  = (const float*)d_in[26];

  // ---- workspace carve ----
  size_t off = 0;
  char* base = (char*)d_ws;
  auto carve = [&](size_t bytes) -> void* {
    void* p = base + off;
    off += (bytes + 255) & ~(size_t)255;
    return p;
  };
  const size_t XPE = (size_t)NBATCH * SEQ * D_MODEL;
  const size_t LTE = (size_t)NBATCH * LLAT * D_MODEL;
  bf16* xp  = (bf16*)carve(XPE * 2);
  bf16* Kc  = (bf16*)carve(XPE * 2);   // cross K, [b][l][d]      (hoisted)
  bf16* Vct = (bf16*)carve(XPE * 2);   // cross V^T, [b][d][l]    (hoisted)
  bf16* lat = (bf16*)carve(LTE * 2);
  bf16* Qb  = (bf16*)carve(LTE * 2);
  bf16* Ksb = (bf16*)carve(LTE * 2);   // self K, [b][l][d]
  bf16* Vst = (bf16*)carve(LTE * 2);   // self V^T, [b][d][l]
  bf16* t0  = (bf16*)carve(LTE * 2);
  bf16* t1  = (bf16*)carve(LTE * 2);
  float* S  = (float*)carve((size_t)NBATCH * LLAT * SEQ * 4);
  bf16*  P  = (bf16*)carve((size_t)NBATCH * LLAT * SEQ * 2);
  // transposed bf16 weights, [N][K]
  bf16* wt[12];
  const int widx[12] = {1, 3, 5, 7, 9, 11, 13, 15, 17, 19, 21, 23};
  for (int i = 0; i < 12; ++i) wt[i] = (bf16*)carve((size_t)D_MODEL * D_MODEL * 2);
  // wt map: 0 cWq 1 cWk 2 cWv 3 cWo 4 sWq 5 sWk 6 sWv 7 sWo 8 d1W1 9 d1W2 10 d2W1 11 d2W2

  const int WN = D_MODEL * D_MODEL;
  for (int i = 0; i < 12; ++i)
    cvt_tr_f32_bf16<<<(WN + 255) / 256, 256, 0, stream>>>((const float*)d_in[widx[i]], wt[i]);

  pe_add_cvt<<<((int)XPE + 255) / 256, 256, 0, stream>>>(x, xp, (int)XPE);
  fill_bf16_zero<<<((int)LTE + 255) / 256, 256, 0, stream>>>(lat, (int)LTE);

  const int   MX  = NBATCH * SEQ;    // 32768
  const int   ML  = NBATCH * LLAT;   // 4096
  const float scl = 1.0f / 32.0f;    // 1/sqrt(1024)
  const long long sQ  = (long long)LLAT * D_MODEL;
  const long long sKV = (long long)SEQ * D_MODEL;
  const int   D = D_MODEL;

  // ---- hoisted cross K, V^T (loop-invariant across DEPTH) ----
  gemm(stream, xp, wt[1], cbk, Kc, nullptr, MX, D, D, D, D, D, 0, 0, 0, 1, 1.0f, 0, 0);
  gemm(stream, xp, wt[2], cbv, Vct, nullptr, SEQ, D, D, D, D, SEQ,
       sKV, 0, sKV, NBATCH, 1.0f, 0, 1);   // V^T: [d][l], ldc = SEQ

  for (int depth = 0; depth < 2; ++depth) {
    // ---- cross attention ----
    gemm(stream, lat, wt[0], cbq, Qb, nullptr, ML, D, D, D, D, D, 0, 0, 0, 1, 1.0f, 0, 0);
    gemm(stream, Qb, Kc, nullptr, nullptr, S, LLAT, SEQ, D, D, D, SEQ,
         sQ, sKV, (long long)LLAT * SEQ, NBATCH, scl, 0, 0);   // Q@K^T/32
    softmax_rows<<<NBATCH * LLAT, 256, 0, stream>>>(S, P, SEQ);
    gemm(stream, P, Vct, nullptr, t0, nullptr, LLAT, D, SEQ, SEQ, SEQ, D,
         (long long)LLAT * SEQ, sKV, sQ, NBATCH, 1.0f, 0, 0);  // P@V
    gemm(stream, t0, wt[3], cbo, t1, nullptr, ML, D, D, D, D, D, 0, 0, 0, 1, 1.0f, 0, 0);
    gemm(stream, t1, wt[8], d1b1, t0, nullptr, ML, D, D, D, D, D, 0, 0, 0, 1, 1.0f, 1, 0);
    gemm(stream, t0, wt[9], d1b2, lat, nullptr, ML, D, D, D, D, D, 0, 0, 0, 1, 1.0f, 1, 0);

    // ---- self attention ----
    gemm(stream, lat, wt[4], sbq, Qb, nullptr, ML, D, D, D, D, D, 0, 0, 0, 1, 1.0f, 0, 0);
    gemm(stream, lat, wt[5], sbk, Ksb, nullptr, ML, D, D, D, D, D, 0, 0, 0, 1, 1.0f, 0, 0);
    gemm(stream, lat, wt[6], sbv, Vst, nullptr, LLAT, D, D, D, D, LLAT,
         sQ, 0, sQ, NBATCH, 1.0f, 0, 1);                       // self V^T: [d][l]
    gemm(stream, Qb, Ksb, nullptr, nullptr, S, LLAT, LLAT, D, D, D, LLAT,
         sQ, sQ, (long long)LLAT * LLAT, NBATCH, scl, 0, 0);
    softmax_rows<<<NBATCH * LLAT, 256, 0, stream>>>(S, P, LLAT);
    gemm(stream, P, Vst, nullptr, t0, nullptr, LLAT, D, LLAT, LLAT, LLAT, D,
         (long long)LLAT * LLAT, sQ, sQ, NBATCH, 1.0f, 0, 0);
    gemm(stream, t0, wt[7], sbo, t1, nullptr, ML, D, D, D, D, D, 0, 0, 0, 1, 1.0f, 0, 0);
    gemm(stream, t1, wt[10], d2b1, t0, nullptr, ML, D, D, D, D, D, 0, 0, 0, 1, 1.0f, 1, 0);
    gemm(stream, t0, wt[11], d2b2, lat, nullptr, ML, D, D, D, D, D, 0, 0, 0, 1, 1.0f, 1, 0);
  }

  head_kernel<<<NBATCH, 256, 0, stream>>>(lat, We, be, (float*)d_out);
}
